// RobustAttentionBlock_67388036874613
// MI455X (gfx1250) — compile-verified, matrix operand order checked by reading, not run on hardware
//
#include <hip/hip_runtime.h>

// ---------------------------------------------------------------------------
// RobustAttentionBlock on MI455X (gfx1250, wave32, WMMA bf16 + pipelined TDM)
//
// Heavy math mapped onto v_wmma_f32_16x16x32_bf16; tile staging done by the
// Tensor Data Mover (tensor_load_to_lds + s_wait_tensorcnt), double-buffered
// so the DMA of tile i+1 overlaps the WMMA work on tile i.
//
//   1) x_norm = LayerNorm(x)                          (bf16 out)
//   2) qkv    = x_norm @ W_qkv                        (GEMM 8192x975x1280)
//   3) q,k,v  = qkv_slice @ W_{q,k,v}                 (GEMM 8192x1280x325, x3)
//   4) *_fft  = C @ *     C[k,n]=cos(2pi kn/S)        (GEMM 2048x1280x2048, z=4)
//      (real part of fft of a real signal == cosine transform; same for ifft)
//   5) scores = q_fft @ k_fft^T                       (GEMM-NT 2048x2048x1280)
//   6) attn   = softmax(|scores|/sqrt(160))           (rowwise, bf16 out)
//   7) out_f  = attn @ v_fft                          (GEMM 2048x1280x2048)
//   8) a_out  = (1/S) C @ out_f                       (GEMM, z=4)
//   9) d_out  = x + a_out @ W_out                     (GEMM w/ fused residual)
// ---------------------------------------------------------------------------

typedef __attribute__((ext_vector_type(16))) __bf16 bf16x16;
typedef __attribute__((ext_vector_type(8)))  float  f32x8;
typedef __attribute__((ext_vector_type(4)))  unsigned int u32x4;
typedef __attribute__((ext_vector_type(8)))  int i32x8;
typedef __attribute__((ext_vector_type(4)))  int i32x4;
typedef unsigned short us;

union FragU { bf16x16 v; unsigned int u[8]; us h[16]; };

__device__ __forceinline__ us f2bf(float f) {
  unsigned int u = __float_as_uint(f);
  u += 0x7FFFu + ((u >> 16) & 1u);   // round-to-nearest-even
  return (us)(u >> 16);
}

#define BM 128
#define BN 128
#define BK 32
#define LDT 40      // A / B^T tile pitch: BK + 8 ushorts -> 80B rows (pad via TDM)
#define LDB_NN 136  // NN B tile pitch:    BN + 8 ushorts -> 272B rows

#if defined(__has_builtin)
#if __has_builtin(__builtin_amdgcn_tensor_load_to_lds) && \
    __has_builtin(__builtin_amdgcn_s_wait_tensorcnt)
#define HAVE_TDM 1
#endif
#endif
#ifndef HAVE_TDM
#define HAVE_TDM 0
#endif

#if HAVE_TDM
// ---------------------------------------------------------------------------
// Issue one TDM 2D tile load: global (strided rows, 2B elems) -> LDS.
// D# per CDNA5 ISA ch.8: group0 = {count, lds_addr, global_addr, type=2},
// group1 = {data_size=2B, pad cfg, tensor_dim0/1 (OOB zero-fill), tile_dim0/1,
// tensor_dim0_stride}. Remaining groups zero (2D tensor). Tracked by TENSORcnt.
//   padi: LDS padding inserted after (1<<padi) 8-byte units stored;
//   pada: (pada+1) DWORDs of LDS padding per insertion.
// 6-arg builtin: (u32x4 g0, i32x8 g1, i32x4 g2, i32x4 g3, i32x8 g4, imm cpol)
// ---------------------------------------------------------------------------
__device__ __forceinline__ void tdm_load_2d(unsigned lds_off, const us* g,
                                            int td0, int td1, int strideElems,
                                            int tile0, int tile1,
                                            int padi, int pada)
{
  unsigned long long ga = (unsigned long long)(size_t)g;
  u32x4 g0 = {0u, 0u, 0u, 0u};
  g0[0] = 1u;                                   // count=1 (valid user D#)
  g0[1] = lds_off;                              // lds_addr (bytes)
  g0[2] = (unsigned)ga;                         // global_addr[31:0]
  g0[3] = (unsigned)(ga >> 32) | (2u << 30);    // global_addr[56:32] | type=2
  i32x8 g1 = {0, 0, 0, 0, 0, 0, 0, 0};
  g1[0] = (1 << 16)                             // data_size: 2 bytes
        | (1 << 20)                             // pad_enable
        | (padi << 22) | (pada << 25);          // pad_interval / pad_amount
  g1[1] = (td0 & 0xFFFF) << 16;                 // tensor_dim0[15:0]
  g1[2] = (int)(((unsigned)td0 >> 16) | (((unsigned)td1 & 0xFFFFu) << 16));
  g1[3] = (int)(((unsigned)td1 >> 16) | ((unsigned)tile0 << 16));  // tile_dim0
  g1[4] = tile1 & 0xFFFF;                       // tile_dim1 (tile_dim2=0)
  g1[5] = strideElems;                          // tensor_dim0_stride[31:0]
  i32x4 z4 = {0, 0, 0, 0};
  i32x8 z8 = {0, 0, 0, 0, 0, 0, 0, 0};
  __builtin_amdgcn_tensor_load_to_lds(g0, g1, z4, z4, z8, 0);
}
#endif

// ---------------------------------------------------------------------------
// Tiled WMMA GEMM: C[MxN] (f32 or bf16) = alpha * A[MxK] * op(B) (+ residual)
//   BT=false: B is KxN row-major (LDS tile kept [k][n]).
//   BT=true : B is NxK row-major (C = A*B^T; LDS tile [n][k], same as A).
// 256 threads = 8 waves (2 in M x 4 in N); wave tile 64x32 = 4x2 WMMA tiles.
// TDM path: double-buffered tiles; wave 0 DMAs tile i+1 while all compute i.
// ---------------------------------------------------------------------------
template<bool BT, bool OUT_BF16>
__global__ __launch_bounds__(256)
void gemm_wmma(const us* __restrict__ A, const us* __restrict__ B,
               void* __restrict__ Cv, const float* __restrict__ residual,
               int M, int N, int K, int lda, int ldb, int ldc,
               long long sA, long long sB, long long sC, float alpha)
{
  __shared__ us ldsA[2][BM * LDT];
  __shared__ us ldsB[2][BT ? (BN * LDT) : (BK * LDB_NN)];

  const int tid   = threadIdx.x;
  const int lane  = tid & 31;
  const int waveM = (tid >> 5) & 1;
  const int waveN = (tid >> 5) >> 1;
  const int m0 = blockIdx.y * BM;
  const int n0 = blockIdx.x * BN;
  const int z  = blockIdx.z;

  A += (size_t)z * (size_t)sA;
  B += (size_t)z * (size_t)sB;

  f32x8 acc[4][2];
#pragma unroll
  for (int i = 0; i < 4; ++i)
#pragma unroll
    for (int j = 0; j < 2; ++j)
#pragma unroll
      for (int e = 0; e < 8; ++e) acc[i][j][e] = 0.0f;

  // ISA 7.12.2: 16-bit operand layout. lane<16 -> row=lane,  K in {0..7,16..23}
  //             lane>=16 -> row=lane-16, K in {8..15,24..31}
  const int rowL  = lane & 15;
  const int kBase = (lane >> 4) * 8;

  const int nk = (K + BK - 1) / BK;

#if HAVE_TDM
  auto stage_tdm = [&](int k0, int buf) {
    const unsigned aoff = (unsigned)(size_t)(void*)&ldsA[buf][0];
    const unsigned boff = (unsigned)(size_t)(void*)&ldsB[buf][0];
    // A tile: BMxBK rows, 80B LDS pitch (pad 64B->+16B: padi=3,pada=3)
    tdm_load_2d(aoff, A + (size_t)m0 * lda + k0,
                K - k0, M - m0, lda, BK, BM, 3, 3);
    if (BT)     // B^T tile: BNxBK rows of B, same pitch as A
      tdm_load_2d(boff, B + (size_t)n0 * ldb + k0,
                  K - k0, N - n0, ldb, BK, BN, 3, 3);
    else        // B tile: BKxBN, 272B pitch (pad 256B->+16B)
      tdm_load_2d(boff, B + (size_t)k0 * ldb + n0,
                  N - n0, K - k0, ldb, BN, BK, 5, 3);
  };
  if (tid < 32) {                     // prologue: tile 0 into buffer 0
    stage_tdm(0, 0);
    __builtin_amdgcn_s_wait_tensorcnt(0);
  }
  __syncthreads();
#endif

  for (int ik = 0; ik < nk; ++ik) {
    const int k0 = ik * BK;
#if HAVE_TDM
    const int cur = ik & 1;
    if (ik + 1 < nk && tid < 32)      // async: DMA next tile into other buffer
      stage_tdm(k0 + BK, cur ^ 1);
#else
    const int cur = 0;
    { // stage A tile [BM][BK] (zero-fill OOB: handles K=325 etc.)
      int m  = tid >> 1;
      int kc = (tid & 1) * 16;
      int gm = m0 + m;
#pragma unroll
      for (int i = 0; i < 16; ++i) {
        int k = kc + i, gk = k0 + k;
        us v = (gm < M && gk < K) ? A[(size_t)gm * lda + gk] : (us)0;
        ldsA[0][m * LDT + k] = v;
      }
    }
    if (BT) {
      int n  = tid >> 1;
      int kc = (tid & 1) * 16;
      int gn = n0 + n;
#pragma unroll
      for (int i = 0; i < 16; ++i) {
        int k = kc + i, gk = k0 + k;
        us v = (gn < N && gk < K) ? B[(size_t)gn * ldb + gk] : (us)0;
        ldsB[0][n * LDT + k] = v;
      }
    } else {
      int kk = tid >> 3, nc = (tid & 7) * 16;
      int gk = k0 + kk;
#pragma unroll
      for (int i = 0; i < 16; ++i) {
        int n = nc + i, gn = n0 + n;
        us v = (gk < K && gn < N) ? B[(size_t)gk * ldb + gn] : (us)0;
        ldsB[0][kk * LDB_NN + n] = v;
      }
    }
    __syncthreads();
#endif

    const us* bA = &ldsA[cur][0];
    const us* bB = &ldsB[cur][0];

    FragU aF[4], bF[2];
#pragma unroll
    for (int mi = 0; mi < 4; ++mi) {
      const us* p = &bA[(waveM * 64 + mi * 16 + rowL) * LDT + kBase];
      const unsigned int* p32 = (const unsigned int*)p;        // K kBase..+7
      const unsigned int* q32 = (const unsigned int*)(p + 16); // K kBase+16..+23
#pragma unroll
      for (int d = 0; d < 4; ++d) { aF[mi].u[d] = p32[d]; aF[mi].u[4 + d] = q32[d]; }
    }
    if (BT) {
#pragma unroll
      for (int ni = 0; ni < 2; ++ni) {
        const us* p = &bB[(waveN * 32 + ni * 16 + rowL) * LDT + kBase];
        const unsigned int* p32 = (const unsigned int*)p;
        const unsigned int* q32 = (const unsigned int*)(p + 16);
#pragma unroll
        for (int d = 0; d < 4; ++d) { bF[ni].u[d] = p32[d]; bF[ni].u[4 + d] = q32[d]; }
      }
    } else {
#pragma unroll
      for (int ni = 0; ni < 2; ++ni) {
        const int ncol = waveN * 32 + ni * 16 + rowL;
#pragma unroll
        for (int j = 0; j < 16; ++j) {
          const int k = kBase + j + ((j >= 8) ? 8 : 0);
          bF[ni].h[j] = bB[k * LDB_NN + ncol];
        }
      }
    }

#pragma unroll
    for (int mi = 0; mi < 4; ++mi)
#pragma unroll
      for (int ni = 0; ni < 2; ++ni)
        acc[mi][ni] = __builtin_amdgcn_wmma_f32_16x16x32_bf16(
            false, aF[mi].v, false, bF[ni].v, (short)0, acc[mi][ni], false, false);

#if HAVE_TDM
    if (tid < 32) __builtin_amdgcn_s_wait_tensorcnt(0);  // next tile resident
    __syncthreads();   // flip buffers: all reads of 'cur' done, 'next' visible
#else
    __syncthreads();
#endif
  }

  // Epilogue. C/D layout: lane col = lane&15, VGPR r -> row r + 8*(lane>>4).
  const int colL = lane & 15;
  const int rowH = (lane >> 4) * 8;
#pragma unroll
  for (int mi = 0; mi < 4; ++mi) {
#pragma unroll
    for (int ni = 0; ni < 2; ++ni) {
      int col = n0 + waveN * 32 + ni * 16 + colL;
      if (col >= N) continue;
      int rowBase = m0 + waveM * 64 + mi * 16 + rowH;
#pragma unroll
      for (int r = 0; r < 8; ++r) {
        int row = rowBase + r;
        if (row >= M) continue;
        float vv = acc[mi][ni][r] * alpha;
        size_t idx = (size_t)z * (size_t)sC + (size_t)row * ldc + col;
        if (OUT_BF16) {
          ((us*)Cv)[idx] = f2bf(vv);
        } else {
          if (residual) vv += residual[(size_t)row * ldc + col];
          ((float*)Cv)[idx] = vv;
        }
      }
    }
  }
}

// ---------------------------------------------------------------------------
// LayerNorm over last dim (E=1280) -> bf16
// ---------------------------------------------------------------------------
__global__ __launch_bounds__(256)
void layernorm_bf16(const float* __restrict__ x, const float* __restrict__ gamma,
                    const float* __restrict__ beta, us* __restrict__ out, int E)
{
  __shared__ float red[2][8];
  const int row = blockIdx.x;
  const float* xr = x + (size_t)row * E;
  float s = 0.f, s2 = 0.f;
  for (int c = threadIdx.x; c < E; c += 256) { float v = xr[c]; s += v; s2 += v * v; }
  for (int o = 16; o > 0; o >>= 1) { s += __shfl_xor(s, o, 32); s2 += __shfl_xor(s2, o, 32); }
  const int lane = threadIdx.x & 31, wave = threadIdx.x >> 5;
  if (lane == 0) { red[0][wave] = s; red[1][wave] = s2; }
  __syncthreads();
  if (wave == 0) {
    s  = (lane < 8) ? red[0][lane] : 0.f;
    s2 = (lane < 8) ? red[1][lane] : 0.f;
    for (int o = 4; o > 0; o >>= 1) { s += __shfl_xor(s, o, 32); s2 += __shfl_xor(s2, o, 32); }
    if (lane == 0) { red[0][0] = s; red[1][0] = s2; }
  }
  __syncthreads();
  const float mu  = red[0][0] / (float)E;
  const float var = red[1][0] / (float)E - mu * mu;
  const float rs  = rsqrtf(var + 1e-5f);
  for (int c = threadIdx.x; c < E; c += 256)
    out[(size_t)row * E + c] = f2bf((xr[c] - mu) * rs * gamma[c] + beta[c]);
}

// ---------------------------------------------------------------------------
// Row softmax of |s * scale| over N=2048 (8 vals/thread kept in registers)
// ---------------------------------------------------------------------------
__global__ __launch_bounds__(256)
void softmax_abs_bf16(const float* __restrict__ S, us* __restrict__ P, int N, float scale)
{
  __shared__ float red[8];
  const int row = blockIdx.x;
  const float* sr = S + (size_t)row * N;
  float v[8], mx = -1e30f;
#pragma unroll
  for (int i = 0; i < 8; ++i) {
    v[i] = fabsf(sr[threadIdx.x + i * 256] * scale);
    mx = fmaxf(mx, v[i]);
  }
  for (int o = 16; o > 0; o >>= 1) mx = fmaxf(mx, __shfl_xor(mx, o, 32));
  const int lane = threadIdx.x & 31, wave = threadIdx.x >> 5;
  if (lane == 0) red[wave] = mx;
  __syncthreads();
  mx = red[0];
#pragma unroll
  for (int w2 = 1; w2 < 8; ++w2) mx = fmaxf(mx, red[w2]);
  float sum = 0.f;
#pragma unroll
  for (int i = 0; i < 8; ++i) { v[i] = __expf(v[i] - mx); sum += v[i]; }
  for (int o = 16; o > 0; o >>= 1) sum += __shfl_xor(sum, o, 32);
  __syncthreads();
  if (lane == 0) red[wave] = sum;
  __syncthreads();
  sum = 0.f;
#pragma unroll
  for (int w2 = 0; w2 < 8; ++w2) sum += red[w2];
  const float inv = 1.0f / sum;
#pragma unroll
  for (int i = 0; i < 8; ++i)
    P[(size_t)row * N + threadIdx.x + i * 256] = f2bf(v[i] * inv);
}

// ---------------------------------------------------------------------------
// Small elementwise kernels
// ---------------------------------------------------------------------------
__global__ void f32_to_bf16_kernel(const float* __restrict__ in, us* __restrict__ out, int n)
{
  int i = blockIdx.x * 256 + threadIdx.x;
  if (i < n) out[i] = f2bf(in[i]);
}

// C[k*Nf + n] = cos(2*pi*k*n/Nf) in bf16 (symmetric; DFT and iDFT real parts)
__global__ void gen_dft_cos(us* __restrict__ C, int Nf)
{
  size_t idx = (size_t)blockIdx.x * 256 + threadIdx.x;
  if (idx >= (size_t)Nf * Nf) return;
  int k = (int)(idx / (size_t)Nf), n = (int)(idx % (size_t)Nf);
  int t = (k * n) & (Nf - 1);                       // exact phase mod Nf
  C[idx] = f2bf(__cosf((float)t * (6.2831853071795864f / (float)Nf)));
}

// qkv[rows x 3R] f32 -> three bf16 [rows x R] slices
__global__ void split_qkv_bf16(const float* __restrict__ qkv, us* __restrict__ q,
                               us* __restrict__ k, us* __restrict__ v, int rows, int R)
{
  size_t i = (size_t)blockIdx.x * 256 + threadIdx.x;
  if (i >= (size_t)rows * R) return;
  int r = (int)(i / (size_t)R), c = (int)(i % (size_t)R);
  const float* p = qkv + (size_t)r * (3 * R);
  q[i] = f2bf(p[c]);
  k[i] = f2bf(p[c + R]);
  v[i] = f2bf(p[c + 2 * R]);
}

// ---------------------------------------------------------------------------
extern "C" void kernel_launch(void* const* d_in, const int* in_sizes, int n_in,
                              void* d_out, int out_size, void* d_ws, size_t ws_size,
                              hipStream_t stream)
{
  (void)in_sizes; (void)n_in; (void)out_size; (void)ws_size;
  const int B = 4, S = 2048, E = 1280, R = 325, R3 = 975;
  const int BS = B * S;                       // 8192 rows
  const long long SE = (long long)S * E;      // per-batch stride

  const float* x    = (const float*)d_in[0];
  const float* Wqkv = (const float*)d_in[1];
  const float* Wq   = (const float*)d_in[2];
  const float* Wk   = (const float*)d_in[3];
  const float* Wv   = (const float*)d_in[4];
  const float* Wout = (const float*)d_in[5];
  const float* gam  = (const float*)d_in[6];
  const float* bet  = (const float*)d_in[7];

  char* wp = (char*)d_ws;
  auto alloc = [&](size_t bytes) -> void* {
    void* p = (void*)wp; wp += (bytes + 255) & ~(size_t)255; return p;
  };
  us* xn    = (us*)alloc((size_t)BS * E * 2);
  us* Wqkvb = (us*)alloc((size_t)E * R3 * 2);
  us* Wqb   = (us*)alloc((size_t)R * E * 2);
  us* Wkb   = (us*)alloc((size_t)R * E * 2);
  us* Wvb   = (us*)alloc((size_t)R * E * 2);
  us* Woutb = (us*)alloc((size_t)E * E * 2);
  us* Cmat  = (us*)alloc((size_t)S * S * 2);
  float* qkv = (float*)alloc((size_t)BS * R3 * 4);
  us* qin = (us*)alloc((size_t)BS * R * 2);
  us* kin = (us*)alloc((size_t)BS * R * 2);
  us* vin = (us*)alloc((size_t)BS * R * 2);
  us* qpr = (us*)alloc((size_t)BS * E * 2);
  us* kpr = (us*)alloc((size_t)BS * E * 2);
  us* vpr = (us*)alloc((size_t)BS * E * 2);
  us* qf  = (us*)alloc((size_t)BS * E * 2);
  us* kf  = (us*)alloc((size_t)BS * E * 2);
  us* vf  = (us*)alloc((size_t)BS * E * 2);
  float* sc = (float*)alloc((size_t)S * S * 4);   // reused across batches
  us* at  = (us*)alloc((size_t)S * S * 2);        // reused across batches
  us* of  = (us*)alloc((size_t)BS * E * 2);
  us* ao  = (us*)alloc((size_t)BS * E * 2);

  auto cdiv = [](int a, int b) { return (a + b - 1) / b; };

  // --- weight / constant prep -------------------------------------------
  f32_to_bf16_kernel<<<cdiv(E * R3, 256), 256, 0, stream>>>(Wqkv, Wqkvb, E * R3);
  f32_to_bf16_kernel<<<cdiv(R * E, 256), 256, 0, stream>>>(Wq, Wqb, R * E);
  f32_to_bf16_kernel<<<cdiv(R * E, 256), 256, 0, stream>>>(Wk, Wkb, R * E);
  f32_to_bf16_kernel<<<cdiv(R * E, 256), 256, 0, stream>>>(Wv, Wvb, R * E);
  f32_to_bf16_kernel<<<cdiv(E * E, 256), 256, 0, stream>>>(Wout, Woutb, E * E);
  gen_dft_cos<<<cdiv(S * S, 256), 256, 0, stream>>>(Cmat, S);

  // --- 1) LayerNorm ------------------------------------------------------
  layernorm_bf16<<<BS, 256, 0, stream>>>(x, gam, bet, xn, E);

  // --- 2) qkv = x_norm @ W_qkv  [8192 x 975 x 1280] ----------------------
  gemm_wmma<false, false><<<dim3(cdiv(R3, BN), cdiv(BS, BM), 1), 256, 0, stream>>>(
      xn, Wqkvb, qkv, nullptr, BS, R3, E, E, R3, R3, 0, 0, 0, 1.0f);
  split_qkv_bf16<<<cdiv(BS * R, 256), 256, 0, stream>>>(qkv, qin, kin, vin, BS, R);

  // --- 3) q/k/v = slice @ W_{q,k,v}  [8192 x 1280 x 325] -----------------
  gemm_wmma<false, true><<<dim3(cdiv(E, BN), cdiv(BS, BM), 1), 256, 0, stream>>>(
      qin, Wqb, qpr, nullptr, BS, E, R, R, E, E, 0, 0, 0, 1.0f);
  gemm_wmma<false, true><<<dim3(cdiv(E, BN), cdiv(BS, BM), 1), 256, 0, stream>>>(
      kin, Wkb, kpr, nullptr, BS, E, R, R, E, E, 0, 0, 0, 1.0f);
  gemm_wmma<false, true><<<dim3(cdiv(E, BN), cdiv(BS, BM), 1), 256, 0, stream>>>(
      vin, Wvb, vpr, nullptr, BS, E, R, R, E, E, 0, 0, 0, 1.0f);

  // --- 4) forward cosine transform (fft real part), batched over z=4 -----
  gemm_wmma<false, true><<<dim3(cdiv(E, BN), cdiv(S, BM), B), 256, 0, stream>>>(
      Cmat, qpr, qf, nullptr, S, E, S, S, E, E, 0, SE, SE, 1.0f);
  gemm_wmma<false, true><<<dim3(cdiv(E, BN), cdiv(S, BM), B), 256, 0, stream>>>(
      Cmat, kpr, kf, nullptr, S, E, S, S, E, E, 0, SE, SE, 1.0f);
  gemm_wmma<false, true><<<dim3(cdiv(E, BN), cdiv(S, BM), B), 256, 0, stream>>>(
      Cmat, vpr, vf, nullptr, S, E, S, S, E, E, 0, SE, SE, 1.0f);

  // --- 5-7) attention per batch (scores buffer reused) -------------------
  const float inv_sqrt_hd = 0.07905694150420949f;   // 1/sqrt(160)
  for (int b = 0; b < B; ++b) {
    const us* qb = qf + (size_t)b * SE;
    const us* kb = kf + (size_t)b * SE;
    const us* vb = vf + (size_t)b * SE;
    gemm_wmma<true, false><<<dim3(cdiv(S, BN), cdiv(S, BM), 1), 256, 0, stream>>>(
        qb, kb, sc, nullptr, S, S, E, E, E, S, 0, 0, 0, 1.0f);
    softmax_abs_bf16<<<S, 256, 0, stream>>>(sc, at, S, inv_sqrt_hd);
    gemm_wmma<false, true><<<dim3(cdiv(E, BN), cdiv(S, BM), 1), 256, 0, stream>>>(
        at, vb, of + (size_t)b * SE, nullptr, S, E, S, S, E, E, 0, 0, 0, 1.0f);
  }

  // --- 8) inverse cosine transform with 1/S scale, batched z=4 -----------
  gemm_wmma<false, true><<<dim3(cdiv(E, BN), cdiv(S, BM), B), 256, 0, stream>>>(
      Cmat, of, ao, nullptr, S, E, S, S, E, E, 0, SE, SE, 1.0f / (float)S);

  // --- 9) d_out = x + attn_out @ W_out  [8192 x 1280 x 1280] -------------
  gemm_wmma<false, false><<<dim3(cdiv(E, BN), cdiv(BS, BM), 1), 256, 0, stream>>>(
      ao, Woutb, (float*)d_out, x, BS, E, E, E, E, E, 0, 0, 0, 1.0f);
}
